// fuse_model_v1_78477642432616
// MI455X (gfx1250) — matrix operand order repaired and checked by reading, not compiled, not run on hardware
//
#include <hip/hip_runtime.h>

// ---------------------------------------------------------------------------
// CDNA5 (gfx1250) implementation of the fused CNN + point-transformer model.
// GEMM-shaped compute (implicit-GEMM convs, QKV/proj/FFN, flash attention)
// runs on v_wmma_f32_16x16x32_f16 (f16 operands, f32 accumulation).
// Weight/B-operand tiles are staged into LDS by the Tensor Data Mover
// (tensor_load_to_lds), double-buffered with s_wait_tensorcnt, using the TDM
// LDS-pad feature (1 DWORD pad per 32) for a conflict-free 33-float stride.
// ---------------------------------------------------------------------------

typedef _Float16 v16h __attribute__((ext_vector_type(16)));
typedef float    v8f  __attribute__((ext_vector_type(8)));
typedef unsigned int u32;
typedef u32 u32x4 __attribute__((ext_vector_type(4)));
typedef u32 u32x8 __attribute__((ext_vector_type(8)));

#define LRELU_SLOPE 0.01f
#define LN_EPS      1e-5f
#define TILE_PAD_STRIDE 33   // 32 data DWORDs + 1 TDM pad DWORD per row
#define TILE_LDS_FLOATS (TILE_PAD_STRIDE * 32 + 4)

__device__ __forceinline__ float geluf(float x) {
  return 0.5f * x * (1.0f + erff(x * 0.70710678118654752f));
}
__device__ __forceinline__ float red16_max(float x) {
#pragma unroll
  for (int m = 1; m < 16; m <<= 1) x = fmaxf(x, __shfl_xor(x, m, 32));
  return x;
}
__device__ __forceinline__ float red16_sum(float x) {
#pragma unroll
  for (int m = 1; m < 16; m <<= 1) x += __shfl_xor(x, m, 32);
  return x;
}
__device__ __forceinline__ float red32_sum(float x) {
#pragma unroll
  for (int m = 1; m < 32; m <<= 1) x += __shfl_xor(x, m, 32);
  return x;
}
// A-fragment K index for element e (16-bit A 16x32, ISA 7.12.2)
__device__ __forceinline__ int kmapA(int e, int hi) {
  return (e < 8) ? (hi * 8 + e) : (16 + hi * 8 + (e - 8));
}
// Generic LDS pointer -> 32-bit LDS offset (aperture low bits, ISA 10.2)
__device__ __forceinline__ u32 lds_off(const void* p) {
  return (u32)(unsigned long long)p;
}

// ---------------------------------------------------------------------------
// Tensor Data Mover: async 2D tile load global->LDS (D# groups 0/1, 2D form).
// tile_w x tile_h elements of 4B, row stride = stride_elems, zero-fill OOB
// against (ten_w, ten_h). LDS pad: +1 DWORD every 32 DWORDs (row stride 33).
// Tracked by TENSORcnt. Portable inline asm (bypasses toolchain builtin skew).
// ---------------------------------------------------------------------------
__device__ __forceinline__ void tdm_load_tile(u32 lds, const float* g,
                                              u32 tile_w, u32 tile_h,
                                              u32 ten_w, u32 ten_h,
                                              u32 stride_elems) {
  unsigned long long ga = (unsigned long long)g;
  u32x4 g0;
  g0[0] = 1u;                                  // count=1, user descriptor
  g0[1] = lds;                                 // lds_addr [63:32]
  g0[2] = (u32)ga;                             // global_addr [95:64]
  g0[3] = (u32)((ga >> 32) & 0x1FFFFFFu) | (2u << 30);  // addr[56:32]|type=2
  u32x8 g1;
  g1[0] = (2u << 16)        // data_size = 4B
        | (1u << 20)        // pad_enable
        | (4u << 22)        // pad_interval: 2^(4+1)=32 DWORDs
        | (0u << 25);       // pad_amount: 1 DWORD
  g1[1] = (ten_w & 0xFFFFu) << 16;                       // tensor_dim0 lo16
  g1[2] = (ten_w >> 16) | ((ten_h & 0xFFFFu) << 16);     // dim0 hi / dim1 lo
  g1[3] = (ten_h >> 16) | (tile_w << 16);                // dim1 hi / tile_dim0
  g1[4] = tile_h;                                        // tile_dim1 (dim2=0)
  g1[5] = stride_elems;                                  // dim0_stride lo32
  g1[6] = 0u;
  g1[7] = 0u;
  asm volatile("tensor_load_to_lds %0, %1" :: "s"(g0), "s"(g1) : "memory");
}
__device__ __forceinline__ void wait_lds_reads_done() {
  asm volatile("s_wait_dscnt 0x0" ::: "memory");
}

// ---------------------------------------------------------------------------
// Implicit-GEMM 3x3 conv, NCHW / OIHW, fused bias + leaky-ReLU, fused
// two-source channel concat (in1 ch [0,C1), in2 ch [C1,C1+C2)). One wave
// computes 16 pixels x 32 out-channels; weights TDM-staged, double-buffered.
// ---------------------------------------------------------------------------
__global__ __launch_bounds__(32) void conv3x3_wmma_kernel(
    const float* __restrict__ in1, int C1,
    const float* __restrict__ in2, int C2,
    int Hi, int Wi,
    const float* __restrict__ w, const float* __restrict__ bias,
    float* __restrict__ out, int Ho, int Wo, int Cout,
    int stride, int pad) {
  const int lane = threadIdx.x, lo = lane & 15, hi = lane >> 4;
  const int P = Ho * Wo;
  const int p0 = blockIdx.x * 16, co0 = blockIdx.y * 32, b = blockIdx.z;
  const int Cin = C1 + C2;
  const int Ktot = Cin * 9;
  const int niter = (Ktot + 31) / 32;

  __shared__ float sW[2][TILE_LDS_FLOATS];

  const int pA = p0 + lo;
  const int oy = pA / Wo, ox = pA - oy * Wo;
  const int iy0 = oy * stride - pad, ix0 = ox * stride - pad;
  const bool rowok = pA < P;
  const size_t inb1 = (size_t)b * C1 * Hi * Wi;
  const size_t inb2 = (size_t)b * C2 * Hi * Wi;

  // weight tile: [32 out-channels] x [32 K], row stride Ktot
  const float* wbase = w + (size_t)co0 * Ktot;
  const u32 ten_h = (u32)(Cout - co0);
  tdm_load_tile(lds_off(&sW[0][0]), wbase, 32u, 32u, (u32)Ktot, ten_h,
                (u32)Ktot);
  if (niter > 1)
    tdm_load_tile(lds_off(&sW[1][0]), wbase + 32, 32u, 32u,
                  (u32)(Ktot - 32), ten_h, (u32)Ktot);

  v8f acc0 = {}, acc1 = {};
  for (int i = 0; i < niter; i++) {
    const int k0 = i * 32;
    if (i + 1 < niter) __builtin_amdgcn_s_wait_tensorcnt(1);
    else               __builtin_amdgcn_s_wait_tensorcnt(0);
    const float* wb = &sW[i & 1][0];
    v16h b0, b1;
#pragma unroll
    for (int e = 0; e < 16; e++) {
      int kk = kmapA(e, hi);
      b0[e] = (_Float16)wb[lo * TILE_PAD_STRIDE + kk];
      b1[e] = (_Float16)wb[(16 + lo) * TILE_PAD_STRIDE + kk];
    }
    wait_lds_reads_done();
    if (i + 2 < niter)
      tdm_load_tile(lds_off(&sW[i & 1][0]), wbase + (k0 + 64), 32u, 32u,
                    (u32)(Ktot - (k0 + 64)), ten_h, (u32)Ktot);
    v16h a;
#pragma unroll
    for (int e = 0; e < 16; e++) {
      int kA = k0 + kmapA(e, hi);
      float av = 0.f;
      if (rowok && kA < Ktot) {
        int ci = kA / 9, r = kA - ci * 9, ky = r / 3, kx = r - ky * 3;
        int iy = iy0 + ky, ix = ix0 + kx;
        if ((unsigned)iy < (unsigned)Hi && (unsigned)ix < (unsigned)Wi) {
          av = (ci < C1) ? in1[inb1 + ((size_t)ci * Hi + iy) * Wi + ix]
                         : in2[inb2 + ((size_t)(ci - C1) * Hi + iy) * Wi + ix];
        }
      }
      a[e] = (_Float16)av;
    }
    acc0 = __builtin_amdgcn_wmma_f32_16x16x32_f16(false, a, false, b0,
                                                  (short)0, acc0, false, false);
    acc1 = __builtin_amdgcn_wmma_f32_16x16x32_f16(false, a, false, b1,
                                                  (short)0, acc1, false, false);
  }
#pragma unroll
  for (int j = 0; j < 2; j++) {
    int co = co0 + j * 16 + lo;
    if (co < Cout) {
      float bb = bias[co];
      const v8f& acc = j ? acc1 : acc0;
#pragma unroll
      for (int v = 0; v < 8; v++) {
        int row = p0 + v + 8 * hi;
        if (row < P) {
          float o = acc[v] + bb;
          o = (o >= 0.f) ? o : LRELU_SLOPE * o;
          out[((size_t)b * Cout + co) * P + row] = o;
        }
      }
    }
  }
}

// ---------------------------------------------------------------------------
// Generic WMMA GEMM: C[M,N] = epilogue(A[M,K] @ B[K,N] + bias [+ residual]).
// One wave computes 16 x 32; B tiles TDM-staged (32 K-rows x 32 cols),
// double-buffered. act: 0 = none, 2 = exact GELU (before residual add).
// ---------------------------------------------------------------------------
__global__ __launch_bounds__(32) void gemm_wmma_kernel(
    const float* __restrict__ A, const float* __restrict__ Bw,
    const float* __restrict__ bias, const float* __restrict__ res,
    float* __restrict__ C, int M, int N, int K, int act) {
  const int lane = threadIdx.x, lo = lane & 15, hi = lane >> 4;
  const int m0 = blockIdx.x * 16, n0 = blockIdx.y * 32;
  const float* arow = A + (size_t)(m0 + lo) * K;
  const int niter = (K + 31) / 32;

  __shared__ float sB[2][TILE_LDS_FLOATS];

  // B tile: [32 K-rows] x [32 cols], row stride N
  const float* bbase = Bw + n0;
  const u32 ten_w = (u32)(N - n0);
  tdm_load_tile(lds_off(&sB[0][0]), bbase, 32u, 32u, ten_w, (u32)K, (u32)N);
  if (niter > 1)
    tdm_load_tile(lds_off(&sB[1][0]), bbase + (size_t)32 * N, 32u, 32u, ten_w,
                  (u32)(K - 32), (u32)N);

  v8f acc0 = {}, acc1 = {};
  for (int i = 0; i < niter; i++) {
    const int k0 = i * 32;
    if (i + 1 < niter) __builtin_amdgcn_s_wait_tensorcnt(1);
    else               __builtin_amdgcn_s_wait_tensorcnt(0);
    const float* bt = &sB[i & 1][0];
    v16h b0, b1;
#pragma unroll
    for (int e = 0; e < 16; e++) {
      int kk = kmapA(e, hi);
      b0[e] = (_Float16)bt[kk * TILE_PAD_STRIDE + lo];
      b1[e] = (_Float16)bt[kk * TILE_PAD_STRIDE + 16 + lo];
    }
    wait_lds_reads_done();
    if (i + 2 < niter)
      tdm_load_tile(lds_off(&sB[i & 1][0]), bbase + (size_t)(k0 + 64) * N, 32u,
                    32u, ten_w, (u32)(K - (k0 + 64)), (u32)N);
    __builtin_prefetch(arow + k0 + 32, 0, 1);
    v16h a;
#pragma unroll
    for (int e = 0; e < 16; e++) {
      int kA = k0 + kmapA(e, hi);
      a[e] = (kA < K) ? (_Float16)arow[kA] : (_Float16)0.f;
    }
    acc0 = __builtin_amdgcn_wmma_f32_16x16x32_f16(false, a, false, b0,
                                                  (short)0, acc0, false, false);
    acc1 = __builtin_amdgcn_wmma_f32_16x16x32_f16(false, a, false, b1,
                                                  (short)0, acc1, false, false);
  }
#pragma unroll
  for (int j = 0; j < 2; j++) {
    int col = n0 + j * 16 + lo;
    if (col < N) {
      float bb = bias ? bias[col] : 0.f;
      const v8f& acc = j ? acc1 : acc0;
#pragma unroll
      for (int v = 0; v < 8; v++) {
        int row = m0 + v + 8 * hi;
        if (row < M) {
          float o = acc[v] + bb;
          if (act == 2) o = geluf(o);
          if (res) o += res[(size_t)row * N + col];
          C[(size_t)row * N + col] = o;
        }
      }
    }
  }
}

// ---------------------------------------------------------------------------
// Flash attention, one wave per (batch, head, 16-query tile). dh in {16,24},
// zero-padded to K=32 for QK^T. P tile routed C-layout -> A-layout via LDS.
// qkv layout: [B, Ntok, 3*d], Q at +0, K at +d, V at +2d within a row.
// ---------------------------------------------------------------------------
__global__ __launch_bounds__(32) void flash_attn_kernel(
    const float* __restrict__ qkv, const int* __restrict__ mask,
    float* __restrict__ out, int Ntok, int d, int dh) {
  const int lane = threadIdx.x, lo = lane & 15, hi = lane >> 4;
  const int b = blockIdx.z, h = blockIdx.y, q0 = blockIdx.x * 16;
  const size_t rs = (size_t)3 * d;
  const float* Qp = qkv + (size_t)b * Ntok * rs + (size_t)h * dh;
  const float* Kp = Qp + d;
  const float* Vp = Qp + 2 * d;

  v16h qa;
#pragma unroll
  for (int e = 0; e < 16; e++) {
    int f = kmapA(e, hi);
    qa[e] = (f < dh) ? (_Float16)Qp[(size_t)(q0 + lo) * rs + f] : (_Float16)0.f;
  }
  float mrow[8], lrow[8];
#pragma unroll
  for (int v = 0; v < 8; v++) { mrow[v] = -3.0e38f; lrow[v] = 0.f; }
  v8f O0 = {}; v8f O1 = {};
  const float rscale = rsqrtf((float)dh);
  __shared__ float sP[16][17];

  for (int kt = 0; kt < Ntok; kt += 16) {
    v16h kb;
#pragma unroll
    for (int e = 0; e < 16; e++) {
      int f = kmapA(e, hi);
      kb[e] = (f < dh) ? (_Float16)Kp[(size_t)(kt + lo) * rs + f]
                       : (_Float16)0.f;
    }
    v8f z = {};
    v8f S = __builtin_amdgcn_wmma_f32_16x16x32_f16(false, qa, false, kb,
                                                   (short)0, z, false, false);
    const bool masked = (mask[(size_t)b * Ntok + kt + lo] != 1);
#pragma unroll
    for (int v = 0; v < 8; v++) {
      float s = S[v] * rscale;
      if (masked) s = -1.0e9f;
      float cm = red16_max(s);               // row stats stay in 16-lane half
      float nm = fmaxf(mrow[v], cm);
      float fac = expf(mrow[v] - nm);
      float p = expf(s - nm);
      lrow[v] = lrow[v] * fac + red16_sum(p);
      mrow[v] = nm;
      O0[v] *= fac; O1[v] *= fac;
      sP[v + 8 * hi][lo] = p;                // C-layout -> LDS
    }
    __syncthreads();
    v16h pa;                                 // re-read in A-layout
#pragma unroll
    for (int e = 0; e < 16; e++) {
      int kl = kmapA(e, hi);
      pa[e] = (kl < 16) ? (_Float16)sP[lo][kl] : (_Float16)0.f;
    }
    v16h vb0;
#pragma unroll
    for (int e = 0; e < 16; e++) {
      int kl = kmapA(e, hi);
      vb0[e] = (kl < 16) ? (_Float16)Vp[(size_t)(kt + kl) * rs + lo]
                         : (_Float16)0.f;
    }
    O0 = __builtin_amdgcn_wmma_f32_16x16x32_f16(false, pa, false, vb0,
                                                (short)0, O0, false, false);
    if (dh > 16) {
      v16h vb1;
#pragma unroll
      for (int e = 0; e < 16; e++) {
        int kl = kmapA(e, hi);
        int f = 16 + lo;
        vb1[e] = (kl < 16 && f < dh) ? (_Float16)Vp[(size_t)(kt + kl) * rs + f]
                                     : (_Float16)0.f;
      }
      O1 = __builtin_amdgcn_wmma_f32_16x16x32_f16(false, pa, false, vb1,
                                                  (short)0, O1, false, false);
    }
    __syncthreads();
  }
#pragma unroll
  for (int v = 0; v < 8; v++) {
    int r = q0 + v + 8 * hi;
    float inv = 1.0f / lrow[v];
    size_t ob = ((size_t)b * Ntok + r) * d + (size_t)h * dh;
    out[ob + lo] = O0[v] * inv;
    if (dh > 16 && (16 + lo) < dh) out[ob + 16 + lo] = O1[v] * inv;
  }
}

// ---------------------------------------------------------------------------
// LayerNorm: one wave per token row.
// ---------------------------------------------------------------------------
__global__ __launch_bounds__(32) void ln_kernel(
    const float* __restrict__ x, const float* __restrict__ g,
    const float* __restrict__ bb, float* __restrict__ y, int d) {
  const int t = blockIdx.x, lane = threadIdx.x;
  const float* xr = x + (size_t)t * d;
  float s = 0.f, s2 = 0.f;
  for (int i = lane; i < d; i += 32) { float v = xr[i]; s += v; s2 += v * v; }
  s = red32_sum(s); s2 = red32_sum(s2);
  float m = s / d, var = s2 / d - m * m;
  float inv = rsqrtf(var + LN_EPS);
  float* yr = y + (size_t)t * d;
  for (int i = lane; i < d; i += 32) yr[i] = (xr[i] - m) * inv * g[i] + bb[i];
}

// align_corners bilinear upsample (separable == einsum with _up_mat twice)
__global__ void upsample_ac_kernel(const float* __restrict__ in,
                                   float* __restrict__ out, int BC,
                                   int Hi, int Wi, int Ho, int Wo) {
  int i = blockIdx.x * blockDim.x + threadIdx.x;
  int total = BC * Ho * Wo;
  if (i >= total) return;
  int px = i % Wo, t = i / Wo, py = t % Ho, bc = t / Ho;
  float fy = py * ((float)(Hi - 1) / (float)(Ho - 1));
  float fx = px * ((float)(Wi - 1) / (float)(Wo - 1));
  int ly = (int)floorf(fy); ly = ly < 0 ? 0 : (ly > Hi - 2 ? Hi - 2 : ly);
  int lx = (int)floorf(fx); lx = lx < 0 ? 0 : (lx > Wi - 2 ? Wi - 2 : lx);
  float wy = fy - ly, wx = fx - lx;
  const float* ip = in + (size_t)bc * Hi * Wi;
  float v00 = ip[(size_t)ly * Wi + lx], v01 = ip[(size_t)ly * Wi + lx + 1];
  float v10 = ip[(size_t)(ly + 1) * Wi + lx], v11 = ip[(size_t)(ly + 1) * Wi + lx + 1];
  out[i] = (1.f - wy) * ((1.f - wx) * v00 + wx * v01) +
           wy * ((1.f - wx) * v10 + wx * v11);
}

// bilinear point-feature extraction (matches _bilinear_extract)
__global__ void extract_kernel(const float* __restrict__ feat,
                               const float* __restrict__ coords,
                               const int* __restrict__ mask,
                               float* __restrict__ ext, int B, int N, int C,
                               int Hf, int Wf, float sf) {
  int t = blockIdx.x * blockDim.x + threadIdx.x;
  if (t >= B * N) return;
  int b = t / N;
  float cx = coords[(size_t)t * 2 + 0] / sf;
  float cy = coords[(size_t)t * 2 + 1] / sf;
  float xl = floorf(cx), xh = ceilf(cx); if (xh == xl) xh += 1.f;
  float yl = floorf(cy), yh = ceilf(cy); if (yh == yl) yh += 1.f;
  xl = fminf(fmaxf(xl, 0.f), (float)(Hf - 2));
  xh = fminf(fmaxf(xh, 1.f), (float)(Hf - 1));
  yl = fminf(fmaxf(yl, 0.f), (float)(Wf - 2));
  yh = fminf(fmaxf(yh, 1.f), (float)(Wf - 1));
  int xli = (int)xl, xhi = (int)xh, yli = (int)yl, yhi = (int)yh;
  float dhx = xh - cx, dxl = cx - xl, dhy = yh - cy, dyl = cy - yl;
  float km = (mask[t] == 1) ? 1.f : 0.f;
  const float* fb = feat + (size_t)b * C * Hf * Wf;
  for (int c = 0; c < C; c++) {
    const float* fc = fb + (size_t)c * Hf * Wf;
    float f_ll = fc[(size_t)xli * Wf + yli], f_lh = fc[(size_t)xli * Wf + yhi];
    float f_hl = fc[(size_t)xhi * Wf + yli], f_hh = fc[(size_t)xhi * Wf + yhi];
    float o = dyl * (dhx * f_lh + dxl * f_hh) + dhy * (dhx * f_ll + dxl * f_hl);
    ext[(size_t)t * C + c] = o * km;
  }
}

// positional MLP: pv = gelu(ln(dt@w1+b1)) @ w2 + b2   (dims 2->2->32)
__global__ void posenc_kernel(const float* __restrict__ dt,
                              const float* __restrict__ w1,
                              const float* __restrict__ b1,
                              const float* __restrict__ g,
                              const float* __restrict__ bb,
                              const float* __restrict__ w2,
                              const float* __restrict__ b2,
                              float* __restrict__ pv, int T) {
  int t = blockIdx.x * blockDim.x + threadIdx.x;
  if (t >= T) return;
  float d0 = dt[(size_t)t * 2], d1 = dt[(size_t)t * 2 + 1];
  float e0 = d0 * w1[0] + d1 * w1[2] + b1[0];
  float e1 = d0 * w1[1] + d1 * w1[3] + b1[1];
  float m = 0.5f * (e0 + e1);
  float va = 0.5f * ((e0 - m) * (e0 - m) + (e1 - m) * (e1 - m));
  float inv = rsqrtf(va + LN_EPS);
  float g0 = geluf((e0 - m) * inv * g[0] + bb[0]);
  float g1 = geluf((e1 - m) * inv * g[1] + bb[1]);
  for (int j = 0; j < 32; j++)
    pv[(size_t)t * 32 + j] = g0 * w2[j] + g1 * w2[32 + j] + b2[j];
}

// x_new[:, :32] = gelu(ln(x@aw)) + pv ; x_new[:, 32:32+C] = ext
__global__ void align_kernel(const float* __restrict__ xsrc, int din,
                             const float* __restrict__ aw,
                             const float* __restrict__ g,
                             const float* __restrict__ bb,
                             const float* __restrict__ pv,
                             const float* __restrict__ ext, int C,
                             float* __restrict__ xout, int d, int T) {
  int t = blockIdx.x * blockDim.x + threadIdx.x;
  if (t >= T) return;
  const float* xr = xsrc + (size_t)t * din;
  float tmp[32];
  float s = 0.f, s2 = 0.f;
  for (int j = 0; j < 32; j++) {
    float acc = 0.f;
    for (int i = 0; i < din; i++) acc += xr[i] * aw[(size_t)i * 32 + j];
    tmp[j] = acc; s += acc; s2 += acc * acc;
  }
  float m = s / 32.f, va = s2 / 32.f - m * m;
  float inv = rsqrtf(va + LN_EPS);
  float* xo = xout + (size_t)t * d;
  for (int j = 0; j < 32; j++)
    xo[j] = geluf((tmp[j] - m) * inv * g[j] + bb[j]) + pv[(size_t)t * 32 + j];
  for (int c = 0; c < C; c++) xo[32 + c] = ext[(size_t)t * C + c];
}

// head: log_softmax(gelu(ln(x@w1+b1)) @ w2 + b2) over last dim 4
__global__ void head_kernel(const float* __restrict__ x,
                            const float* __restrict__ w1,
                            const float* __restrict__ b1,
                            const float* __restrict__ g,
                            const float* __restrict__ bb,
                            const float* __restrict__ w2,
                            const float* __restrict__ b2,
                            float* __restrict__ out, int T) {
  int t = blockIdx.x * blockDim.x + threadIdx.x;
  if (t >= T) return;
  const float* xr = x + (size_t)t * 64;
  float y[32];
  float s = 0.f, s2 = 0.f;
  for (int j = 0; j < 32; j++) {
    float acc = b1[j];
    for (int i = 0; i < 64; i++) acc += xr[i] * w1[(size_t)i * 32 + j];
    y[j] = acc; s += acc; s2 += acc * acc;
  }
  float m = s / 32.f, va = s2 / 32.f - m * m;
  float inv = rsqrtf(va + LN_EPS);
  for (int j = 0; j < 32; j++) y[j] = geluf((y[j] - m) * inv * g[j] + bb[j]);
  float z[4];
  for (int k = 0; k < 4; k++) {
    float acc = b2[k];
    for (int j = 0; j < 32; j++) acc += y[j] * w2[(size_t)j * 4 + k];
    z[k] = acc;
  }
  float mx = fmaxf(fmaxf(z[0], z[1]), fmaxf(z[2], z[3]));
  float se = expf(z[0] - mx) + expf(z[1] - mx) + expf(z[2] - mx) + expf(z[3] - mx);
  float ls = logf(se);
  for (int k = 0; k < 4; k++) out[(size_t)t * 4 + k] = z[k] - mx - ls;
}

// ---------------------------------------------------------------------------
// Host orchestration
// ---------------------------------------------------------------------------
extern "C" void kernel_launch(void* const* d_in, const int* in_sizes, int n_in,
                              void* d_out, int out_size, void* d_ws,
                              size_t ws_size, hipStream_t stream) {
  (void)in_sizes; (void)n_in; (void)out_size; (void)ws_size;
  const int B = 2, N = 2048;
  const int T = B * N;        // 4096 tokens
  const int M = T;            // GEMM row count (multiple of 16)

  const float* x_in   = (const float*)d_in[0];
  const float* dt     = (const float*)d_in[1];
  const float* dens   = (const float*)d_in[2];
  const int*   mask   = (const int*)d_in[3];
  const float* coords = (const float*)d_in[4];

  // ---- params: jax tree_flatten order (dict keys sorted alphabetically) ----
  int pi = 5;
  auto nxt = [&]() { return (const float*)d_in[pi++]; };
  const float *a_b[3], *a_g[3], *a_w[3];
  for (int i = 0; i < 3; i++) { a_b[i] = nxt(); a_g[i] = nxt(); a_w[i] = nxt(); }
  const float *dc_b[3], *dc_w[3];
  for (int i = 0; i < 3; i++) { dc_b[i] = nxt(); dc_w[i] = nxt(); }
  const float *e_b[7], *e_w[7];
  for (int i = 0; i < 7; i++) { e_b[i] = nxt(); e_w[i] = nxt(); }
  const float *h_b = nxt(), *h_b1 = nxt(), *h_b2 = nxt(), *h_g = nxt(),
              *h_w1 = nxt(), *h_w2 = nxt();
  const float *p_b = nxt(), *p_b1 = nxt(), *p_b2 = nxt(), *p_g = nxt(),
              *p_w1 = nxt(), *p_w2 = nxt();
  struct Blk {
    const float *ffn_b1, *ffn_b2, *ffn_w1, *ffn_w2, *ln1_b, *ln1_g,
                *ln2_b, *ln2_g, *proj_b, *proj_w, *qkv_b, *qkv_w;
  } blk[10];
  for (int i = 0; i < 10; i++) {
    blk[i].ffn_b1 = nxt(); blk[i].ffn_b2 = nxt(); blk[i].ffn_w1 = nxt();
    blk[i].ffn_w2 = nxt(); blk[i].ln1_b = nxt();  blk[i].ln1_g = nxt();
    blk[i].ln2_b = nxt();  blk[i].ln2_g = nxt();  blk[i].proj_b = nxt();
    blk[i].proj_w = nxt(); blk[i].qkv_b = nxt();  blk[i].qkv_w = nxt();
  }

  // ---- workspace arena ----
  char* wsb = (char*)d_ws;
  size_t off = 0;
  auto allocf = [&](size_t nfl) -> float* {
    float* p = (float*)(wsb + off);
    off += ((nfl * 4 + 255) / 256) * 256;
    return p;
  };
  float* f_h0   = allocf((size_t)B * 32 * 384 * 384);
  float* f_t1   = allocf((size_t)B * 64 * 192 * 192);
  float* f_h1   = allocf((size_t)B * 64 * 192 * 192);
  float* f_t2   = allocf((size_t)B * 64 * 64 * 64);
  float* f_h2   = allocf((size_t)B * 64 * 64 * 64);
  float* f_t3   = allocf((size_t)B * 128 * 32 * 32);
  float* f_fe   = allocf((size_t)B * 128 * 32 * 32);
  float* f_up   = allocf((size_t)B * 64 * 384 * 384);   // reused each stage
  float* f_dec0 = allocf((size_t)B * 64 * 64 * 64);
  float* f_dec1 = allocf((size_t)B * 64 * 192 * 192);
  float* f_dec2 = allocf((size_t)B * 32 * 384 * 384);
  float* f_pv   = allocf((size_t)T * 32);
  float* f_ext  = allocf((size_t)T * 64);
  float* f_x0   = allocf((size_t)T * 96);
  float* f_x1   = allocf((size_t)T * 96);
  float* f_ln   = allocf((size_t)T * 96);
  float* f_qkv  = allocf((size_t)T * 288);
  float* f_att  = allocf((size_t)T * 96);
  float* f_ffn  = allocf((size_t)T * 16);

  auto conv = [&](const float* i1, int C1, const float* i2, int C2, int Hi,
                  int Wi, const float* w, const float* bi, float* o, int Ho,
                  int Wo, int Co, int st, int pd) {
    dim3 g((Ho * Wo + 15) / 16, (Co + 31) / 32, B);
    conv3x3_wmma_kernel<<<g, 32, 0, stream>>>(i1, C1, i2, C2, Hi, Wi, w, bi, o,
                                              Ho, Wo, Co, st, pd);
  };
  auto gemm = [&](const float* A, const float* Bw, const float* bi,
                  const float* rs, float* C, int Nn, int K, int act) {
    dim3 g(M / 16, (Nn + 31) / 32);
    gemm_wmma_kernel<<<g, 32, 0, stream>>>(A, Bw, bi, rs, C, M, Nn, K, act);
  };

  // ---- encoder ----
  conv(dens, 1, nullptr, 0, 384, 384, e_w[0], e_b[0], f_h0, 384, 384, 32, 1, 1);
  conv(f_h0, 32, nullptr, 0, 384, 384, e_w[1], e_b[1], f_t1, 192, 192, 64, 2, 1);
  conv(f_t1, 64, nullptr, 0, 192, 192, e_w[2], e_b[2], f_h1, 192, 192, 64, 1, 1);
  conv(f_h1, 64, nullptr, 0, 192, 192, e_w[3], e_b[3], f_t2, 64, 64, 64, 3, 0);
  conv(f_t2, 64, nullptr, 0, 64, 64, e_w[4], e_b[4], f_h2, 64, 64, 64, 1, 1);
  conv(f_h2, 64, nullptr, 0, 64, 64, e_w[5], e_b[5], f_t3, 32, 32, 128, 2, 1);
  conv(f_t3, 128, nullptr, 0, 32, 32, e_w[6], e_b[6], f_fe, 32, 32, 128, 1, 1);

  // ---- positional encoding ----
  posenc_kernel<<<(T + 127) / 128, 128, 0, stream>>>(dt, p_w1, p_b1, p_g, p_b,
                                                     p_w2, p_b2, f_pv, T);

  // ---- stage configs ----
  const float* sk_f[3]  = {f_h2, f_h1, f_h0};
  const int    sk_c[3]  = {64, 64, 32};
  const float* up_in[3] = {f_fe, f_dec0, f_dec1};
  const int    up_c[3]  = {128, 64, 64};
  const int    up_hi[3] = {32, 64, 192};
  const int    up_ho[3] = {64, 192, 384};
  float*       dec_o[3] = {f_dec0, f_dec1, f_dec2};
  const int    dec_c[3] = {64, 64, 32};
  const float  sfv[3]   = {6.f, 2.f, 1.f};
  const int    dstage[3] = {96, 96, 64};
  const int    dinv[3]   = {3, 96, 96};
  const int    blk0[3]   = {0, 3, 6};
  const int    nblk[3]   = {3, 3, 4};

  float* xA = f_x0;
  float* xB = f_x1;

  for (int bi = 0; bi < 3; bi++) {
    const int d = dstage[bi], dh = d / 4, Cx = dec_c[bi];
    // upsample (align_corners) + fused-concat decoder conv + lrelu
    {
      int tot = B * up_c[bi] * up_ho[bi] * up_ho[bi];
      upsample_ac_kernel<<<(tot + 255) / 256, 256, 0, stream>>>(
          up_in[bi], f_up, B * up_c[bi], up_hi[bi], up_hi[bi], up_ho[bi],
          up_ho[bi]);
      conv(sk_f[bi], sk_c[bi], f_up, up_c[bi], up_ho[bi], up_ho[bi], dc_w[bi],
           dc_b[bi], dec_o[bi], up_ho[bi], up_ho[bi], dec_c[bi], 1, 1);
    }
    // bilinear point extraction
    extract_kernel<<<(T + 127) / 128, 128, 0, stream>>>(
        dec_o[bi], coords, mask, f_ext, B, N, Cx, up_ho[bi], up_ho[bi], sfv[bi]);
    // align + concat -> new x
    {
      const float* xsrc = (bi == 0) ? x_in : xA;
      align_kernel<<<(T + 127) / 128, 128, 0, stream>>>(
          xsrc, dinv[bi], a_w[bi], a_g[bi], a_b[bi], f_pv, f_ext, Cx, xB, d, T);
      float* t = xA; xA = xB; xB = t;   // xA now holds current x
    }
    // transformer blocks
    for (int j = 0; j < nblk[bi]; j++) {
      const Blk& p = blk[blk0[bi] + j];
      ln_kernel<<<M, 32, 0, stream>>>(xA, p.ln1_g, p.ln1_b, f_ln, d);
      gemm(f_ln, p.qkv_w, p.qkv_b, nullptr, f_qkv, 3 * d, d, 0);
      {
        dim3 g(N / 16, 4, B);
        flash_attn_kernel<<<g, 32, 0, stream>>>(f_qkv, mask, f_att, N, d, dh);
      }
      gemm(f_att, p.proj_w, p.proj_b, xA, xB, d, d, 0);      // xB = x + attn
      ln_kernel<<<M, 32, 0, stream>>>(xB, p.ln2_g, p.ln2_b, f_ln, d);
      gemm(f_ln, p.ffn_w1, p.ffn_b1, nullptr, f_ffn, 16, d, 2); // gelu
      gemm(f_ffn, p.ffn_w2, p.ffn_b2, xB, xA, d, 16, 0);     // xA = xB + ffn
    }
  }

  // ---- head + log_softmax ----
  head_kernel<<<(T + 127) / 128, 128, 0, stream>>>(
      xA, h_w1, h_b1, h_g, h_b, h_w2, h_b2, (float*)d_out, T);
}
